// scs_GMN_40286793236484
// MI455X (gfx1250) — matrix operand order repaired and checked by reading, not compiled, not run on hardware
//
#include <hip/hip_runtime.h>
#include <hip/hip_bf16.h>

// ---------------------------------------------------------------------------
// GMN forward for MI455X (gfx1250, wave32, WMMA).
//
// Dominant cost: four 10000x10000x256 GEMM-like passes (adj @ X three times, and
// the F F^T row-norm pass). adj (400MB fp32) streams from HBM (~17us/pass at
// 23.3 TB/s); the feature operand is pre-converted to bf16 once (5MB, lives
// in the 192MB L2), so inner-loop staging is pure 16B copies with no cvt
// work. bf16 is exact for the 0/1 adjacency and adequate for N(0,1)
// features; accumulation is fp32 via v_wmma_f32_16x16x32_bf16.
// gcn_agg uses 32x256 block tiles: one X staging feeds 32 WMMAs.
// ---------------------------------------------------------------------------

#define NDA 10000
#define NQ  64
#define NC  1024
#define DIN 128
#define DD  256

typedef __attribute__((ext_vector_type(16))) __bf16 v16bf;
typedef __attribute__((ext_vector_type(8)))  __bf16 bf8;   // 16 bytes
typedef __attribute__((ext_vector_type(8)))  float  v8f;
typedef __attribute__((ext_vector_type(4)))  float  f4;

__device__ __forceinline__ float lrelu(float x) { return x > 0.f ? x : 0.01f * x; }

__device__ __forceinline__ v8f wmma_bf16(v16bf a, v16bf b, v8f c) {
  // (neg_a, A, neg_b, B, c_mod, C, reuse_a, reuse_b)
  return __builtin_amdgcn_wmma_f32_16x16x32_bf16(false, a, false, b, (short)0, c,
                                                 false, false);
}

// A-fragment (16x32 bf16): lane L<16 holds row M=L, K=0..7 & 16..23;
// lane L+16 holds row M=L, K=8..15 & 24..31 (ISA 7.12.2).
// B-fragment (32x16): lane L holds column N=L&15, K=0..7 & 16..23 (L<16), K=8..15 & 24..31 (L>=16).
// C/D: element i of lane L is (M = i + 8*(L>>4), N = L&15).

// ---------------------------------------------------------------------------
// Generic GEMM: C[M,N] = act(A[M,K] @ B[K,N]), fp32 in, fp32 or bf16 out.
// One wave per 16x16 tile. M,N multiples of 16; K multiple of 32.
// A loads are 16B-aligned float4; B is read with scalar loads.
// ---------------------------------------------------------------------------
__global__ __launch_bounds__(256) void gemm_wmma_kernel(
    const float* __restrict__ A, const float* __restrict__ B,
    void* __restrict__ Cv, int M, int N, int K, int act, int obf) {
  int wave = (blockIdx.x * blockDim.x + threadIdx.x) >> 5;
  int lane = threadIdx.x & 31;
  int tn16 = N >> 4;
  int ntiles = (M >> 4) * tn16;
  if (wave >= ntiles) return;            // wave-uniform: EXEC stays full
  int tm = (wave / tn16) << 4;
  int tn = (wave % tn16) << 4;
  int arow = lane & 15;
  int kb = (lane >> 4) << 3;
  const float* Arow = A + (size_t)(tm + arow) * K + kb;
  v8f acc = {};
  for (int kk = 0; kk < K; kk += 32) {
    v16bf a, b;
    {
      const f4* p0 = (const f4*)(Arow + kk);
      const f4* p1 = (const f4*)(Arow + kk + 16);
      f4 x0 = p0[0], x1 = p0[1], y0 = p1[0], y1 = p1[1];
#pragma unroll
      for (int q = 0; q < 4; ++q) {
        a[q]      = (__bf16)x0[q];
        a[4 + q]  = (__bf16)x1[q];
        a[8 + q]  = (__bf16)y0[q];
        a[12 + q] = (__bf16)y1[q];
      }
    }
    {
      const float* bp = B + (size_t)(kk + kb) * N + tn + arow;
#pragma unroll
      for (int q = 0; q < 8; ++q) {
        b[q]     = (__bf16)bp[(size_t)q * N];
        b[8 + q] = (__bf16)bp[(size_t)(16 + q) * N];
      }
    }
    acc = wmma_bf16(a, b, acc);
  }
  int mb = (lane >> 4) << 3, n = lane & 15;
  if (obf) {
    __bf16* C = (__bf16*)Cv;
#pragma unroll
    for (int i = 0; i < 8; ++i) {
      float v = acc[i];
      if (act) v = lrelu(v);
      C[(size_t)(tm + mb + i) * N + tn + n] = (__bf16)v;
    }
  } else {
    float* C = (float*)Cv;
#pragma unroll
    for (int i = 0; i < 8; ++i) {
      float v = acc[i];
      if (act) v = lrelu(v);
      C[(size_t)(tm + mb + i) * N + tn + n] = v;
    }
  }
}

// ---------------------------------------------------------------------------
// GCN aggregation: out[R,256] = act(adj[R,K] @ X[K,256]), X already bf16.
// Block = 256 thr = 8 waves; block owns 32 output rows x all 256 cols:
// wave w -> row-tile (w>>2), cols [64*(w&3), 64*(w&3)+64) = 4 fragments.
// One X staging (32x256) feeds 32 WMMAs. adj staged fp32->bf16; X staged by
// pure 16B copies. Row tail (R % 32 == 16) and K tail (K % 32 == 16)
// zero-filled/guarded; K % 16 == 0 so vectors never straddle the boundary.
// ---------------------------------------------------------------------------
__global__ __launch_bounds__(256) void gcn_agg_kernel(
    const float* __restrict__ adj, const __bf16* __restrict__ X,
    float* __restrict__ out, int R, int K, int act) {
  __shared__ __bf16 sA[32][48];        // 32 x 32 (+16 pad), 96B rows
  __shared__ __bf16 sB[32][DD + 16];   // 32 x 256 (+16 pad), 544B rows
  int t = threadIdx.x;
  int wave = t >> 5, lane = t & 31;
  int row0 = blockIdx.x << 5;          // 32 rows per block
  int rt = (wave >> 2) << 4;           // row-tile base within block (0 or 16)
  int cb = (wave & 3) << 6;            // column base (0,64,128,192)
  int arow = lane & 15, kb = (lane >> 4) << 3;
  v8f acc[4] = {};
  for (int kk = 0; kk < K; kk += 32) {
    {                                   // adj tile: 32x32 = 256 float4
      int r = t >> 3, c4 = t & 7;
      f4 v = {};
      if (row0 + r < R && kk + c4 * 4 < K)
        v = *(const f4*)(adj + (size_t)(row0 + r) * K + kk + c4 * 4);
#pragma unroll
      for (int q = 0; q < 4; ++q) sA[r][c4 * 4 + q] = (__bf16)v[q];
    }
#pragma unroll
    for (int r4 = 0; r4 < 4; ++r4) {    // X tile: 32x256 bf16 = 1024 x 16B
      int idx = t + (r4 << 8);
      int r = idx >> 5, c8 = idx & 31;
      bf8 v = {};
      if (kk + r < K)
        v = *(const bf8*)(X + (size_t)(kk + r) * DD + c8 * 8);
#pragma unroll
      for (int q = 0; q < 8; ++q) sB[r][c8 * 8 + q] = v[q];
    }
    __syncthreads();
    v16bf a;
#pragma unroll
    for (int j = 0; j < 8; ++j) a[j] = sA[rt + arow][kb + j];
#pragma unroll
    for (int j = 0; j < 8; ++j) a[8 + j] = sA[rt + arow][16 + kb + j];
#pragma unroll
    for (int f = 0; f < 4; ++f) {
      v16bf b;
#pragma unroll
      for (int j = 0; j < 8; ++j) {
        b[j]     = sB[kb + j][cb + f * 16 + arow];
        b[8 + j] = sB[16 + kb + j][cb + f * 16 + arow];
      }
      acc[f] = wmma_bf16(a, b, acc[f]);
    }
    __syncthreads();
  }
  int mb = (lane >> 4) << 3, n = lane & 15;
#pragma unroll
  for (int f = 0; f < 4; ++f) {
#pragma unroll
    for (int i = 0; i < 8; ++i) {
      int row = row0 + rt + mb + i;
      float v = acc[f][i];
      if (act) v = lrelu(v);
      if (row < R) out[(size_t)row * DD + cb + f * 16 + n] = v;
    }
  }
}

// ---------------------------------------------------------------------------
// Row sums of squares of re_adj = F F^T without materializing re_adj.
// F is bf16. Block owns 16 rows of F (A-fragments preloaded to registers).
// Outer loop: 256-column chunks; per chunk the 8 k-slabs of F^T are staged
// in LDS via pure 16B copies (thread t copies a contiguous 32-elem segment
// of F row n0+t). Wave w covers columns [32w,32w+32). Per-lane partial row
// sums reduced via shfl_xor, one atomic per row per wave.
// ---------------------------------------------------------------------------
__global__ __launch_bounds__(256) void re_rownorm_kernel(
    const __bf16* __restrict__ F, float* __restrict__ S2, int R) {
  __shared__ __bf16 sBt[256][32];      // [n_local][k_local] 16KB, 64B rows
  int t = threadIdx.x, wave = t >> 5, lane = t & 31;
  int row0 = blockIdx.x << 4;
  int arow = lane & 15, kb = (lane >> 4) << 3;
  int cb = wave << 5;
  const __bf16* Ar = F + (size_t)(row0 + arow) * DD + kb;
  v16bf afr[8];
#pragma unroll
  for (int ks = 0; ks < 8; ++ks) {
    bf8 x0 = *(const bf8*)(Ar + ks * 32);
    bf8 x1 = *(const bf8*)(Ar + ks * 32 + 16);
#pragma unroll
    for (int q = 0; q < 8; ++q) {
      afr[ks][q]     = x0[q];
      afr[ks][8 + q] = x1[q];
    }
  }
  float rowacc[8];
#pragma unroll
  for (int i = 0; i < 8; ++i) rowacc[i] = 0.f;

  for (int n0 = 0; n0 < R; n0 += 256) {
    v8f c0 = {}, c1 = {};
    int gr = n0 + t;
    const bf8* src = (const bf8*)(F + (size_t)gr * DD);
#pragma unroll
    for (int ks = 0; ks < 8; ++ks) {
      // stage F^T slab: sBt[nl][kl] = F[n0+nl][ks*32+kl]
#pragma unroll
      for (int c8 = 0; c8 < 4; ++c8) {
        bf8 v = {};
        if (gr < R) v = src[ks * 4 + c8];
#pragma unroll
        for (int q = 0; q < 8; ++q) sBt[t][c8 * 8 + q] = v[q];
      }
      __syncthreads();
      v16bf b0, b1;
#pragma unroll
      for (int j = 0; j < 8; ++j) { b0[j] = sBt[cb + arow][kb + j]; b0[8 + j] = sBt[cb + arow][16 + kb + j]; }
#pragma unroll
      for (int j = 0; j < 8; ++j) { b1[j] = sBt[cb + 16 + arow][kb + j]; b1[8 + j] = sBt[cb + 16 + arow][16 + kb + j]; }
      c0 = wmma_bf16(afr[ks], b0, c0);
      c1 = wmma_bf16(afr[ks], b1, c1);
      __syncthreads();
    }
#pragma unroll
    for (int i = 0; i < 8; ++i) rowacc[i] += c0[i] * c0[i] + c1[i] * c1[i];
  }
#pragma unroll
  for (int i = 0; i < 8; ++i) {
    float v = rowacc[i];
    v += __shfl_xor(v, 1, 32);
    v += __shfl_xor(v, 2, 32);
    v += __shfl_xor(v, 4, 32);
    v += __shfl_xor(v, 8, 32);
    if ((lane & 15) == 0)
      atomicAdd(&S2[row0 + ((lane >> 4) << 3) + i], v);
  }
}

// --------------------------- small helper kernels ---------------------------

// Per-row gather (optional) + raw copy (optional) + row L2-normalize (eps 1e-12).
__global__ __launch_bounds__(256) void gather_norm_kernel(
    const float* __restrict__ src, const int* __restrict__ cand,
    float* __restrict__ pre, float* __restrict__ nrm) {
  __shared__ float red[256];
  __shared__ float invn;
  int r = blockIdx.x, t = threadIdx.x;
  int s = cand ? cand[r] : r;
  float v = src[(size_t)s * DD + t];
  if (pre) pre[(size_t)r * DD + t] = v;
  red[t] = v * v;
  __syncthreads();
  for (int off = 128; off > 0; off >>= 1) {
    if (t < off) red[t] += red[t + off];
    __syncthreads();
  }
  if (t == 0) invn = 1.f / fmaxf(sqrtf(red[0]), 1e-12f);
  __syncthreads();
  nrm[(size_t)r * DD + t] = v * invn;
}

// danT[k][n] = dan[n][k]; dan is [NC][DD]. Coalesced writes.
__global__ void transpose_kernel(const float* __restrict__ dan,
                                 float* __restrict__ danT, int n) {
  int i = blockIdx.x * blockDim.x + threadIdx.x;
  if (i < n) {
    int k = i >> 10, c = i & (NC - 1);       // i = k*NC + c
    danT[i] = dan[(size_t)c * DD + k];
  }
}

// att_q = lrelu(q + h / max(||h||_col, 1e-12)), column norms over NQ rows.
__global__ __launch_bounds__(256) void col_norm_add_lrelu_kernel(
    const float* __restrict__ q, const float* __restrict__ h,
    float* __restrict__ out) {
  int c = threadIdx.x;
  float ss = 0.f;
  for (int r = 0; r < NQ; ++r) { float v = h[(size_t)r * DD + c]; ss += v * v; }
  float inv = 1.f / fmaxf(sqrtf(ss), 1e-12f);
  for (int r = 0; r < NQ; ++r) {
    size_t idx = (size_t)r * DD + c;
    out[idx] = lrelu(q[idx] + h[idx] * inv);
  }
}

// emb[c] = mean over NQ rows; emb[DD] = ||emb||.
__global__ __launch_bounds__(256) void colmean_kernel(
    const float* __restrict__ aq, float* __restrict__ emb) {
  __shared__ float red[256];
  int c = threadIdx.x;
  float s = 0.f;
  for (int r = 0; r < NQ; ++r) s += aq[(size_t)r * DD + c];
  float mval = s * (1.f / NQ);
  emb[c] = mval;
  red[c] = mval * mval;
  __syncthreads();
  for (int off = 128; off > 0; off >>= 1) {
    if (c < off) red[c] += red[c + off];
    __syncthreads();
  }
  if (c == 0) emb[DD] = sqrtf(red[0]);
}

// end[i] = (attda2[i]·emb) / max(||attda2[i]||*||emb||, 1e-8); msk = end>thr.
__global__ __launch_bounds__(256) void end_mask_kernel(
    const float* __restrict__ attda, const float* __restrict__ emb,
    const float* __restrict__ thr, float* __restrict__ endv,
    float* __restrict__ msk) {
  __shared__ float rd[256], rs[256];
  int i = blockIdx.x, t = threadIdx.x;
  float v = attda[(size_t)i * DD + t];
  rd[t] = v * emb[t];
  rs[t] = v * v;
  __syncthreads();
  for (int off = 128; off > 0; off >>= 1) {
    if (t < off) { rd[t] += rd[t + off]; rs[t] += rs[t + off]; }
    __syncthreads();
  }
  if (t == 0) {
    float den = fmaxf(sqrtf(rs[0]) * emb[DD], 1e-8f);
    float e = rd[0] / den;
    endv[i] = e;
    msk[i] = (e > thr[0]) ? 1.f : 0.f;
  }
}

// out = lrelu(in), float4-vectorized (n4 = n/4 elements).
__global__ void lrelu4_kernel(const f4* __restrict__ in, f4* __restrict__ out,
                              int n4) {
  int i = blockIdx.x * blockDim.x + threadIdx.x;
  if (i < n4) {
    f4 v = in[i];
#pragma unroll
    for (int q = 0; q < 4; ++q) v[q] = lrelu(v[q]);
    out[i] = v;
  }
}

// Fbf = bf16(attda2 * msk[row]); 64 float4 per 256-col row.
__global__ void mask_rows_bf_kernel(const f4* __restrict__ in,
                                    const float* __restrict__ msk,
                                    __bf16* __restrict__ out, int n4) {
  int i = blockIdx.x * blockDim.x + threadIdx.x;
  if (i < n4) {
    float m = msk[i >> 6];
    f4 v = in[i];
#pragma unroll
    for (int q = 0; q < 4; ++q) out[(size_t)i * 4 + q] = (__bf16)(v[q] * m);
  }
}

// SA[i] = F[i]·G[i]  (= sum_j re_adj[i,j]*adj[i,j] with G = adj@F)
// DV[i] = ||F[i]||^2 * adj[i,i]  (= re_adj[i,i]*adj[i,i])
__global__ __launch_bounds__(256) void row_dot_kernel(
    const __bf16* __restrict__ F, const float* __restrict__ G,
    const float* __restrict__ adj, float* __restrict__ SA,
    float* __restrict__ DV) {
  __shared__ float rd[256], rs[256];
  int i = blockIdx.x, t = threadIdx.x;
  float f = (float)F[(size_t)i * DD + t];
  float g = G[(size_t)i * DD + t];
  rd[t] = f * g;
  rs[t] = f * f;
  __syncthreads();
  for (int off = 128; off > 0; off >>= 1) {
    if (t < off) { rd[t] += rd[t + off]; rs[t] += rs[t + off]; }
    __syncthreads();
  }
  if (t == 0) {
    SA[i] = rd[0];
    DV[i] = rs[0] * adj[(size_t)i * NDA + i];
  }
}

__global__ __launch_bounds__(256) void finalize_kernel(
    const float* __restrict__ S2, const float* __restrict__ SA,
    const float* __restrict__ DV, const float* __restrict__ msk,
    float* __restrict__ outs) {
  __shared__ float rc[256], rt[256], rr[256];
  int t = threadIdx.x;
  float cnt = 0.f, tot = 0.f, tr = 0.f;
  for (int i = t; i < NDA; i += 256) {
    cnt += msk[i];
    float rn = fmaxf(sqrtf(S2[i]), 1e-12f);
    tot += SA[i] / rn;
    tr  += DV[i] / rn;
  }
  rc[t] = cnt; rt[t] = tot; rr[t] = tr;
  __syncthreads();
  for (int off = 128; off > 0; off >>= 1) {
    if (t < off) { rc[t] += rc[t + off]; rt[t] += rt[t + off]; rr[t] += rr[t + off]; }
    __syncthreads();
  }
  if (t == 0) {
    float c = rc[0], T = rt[0], R = rr[0];
    bool has = c > 0.f;
    outs[0] = has ? T / fmaxf(c, 1.f) : 0.f;                 // pre_avg_degree
    outs[1] = has ? 2.f * T / (R * (R - 1.f) + 1e-4f) : 0.f; // pre_density
    outs[2] = has ? R : 0.f;                                 // pre_avg_nodes
  }
}

// ---------------------------------------------------------------------------
extern "C" void kernel_launch(void* const* d_in, const int* in_sizes, int n_in,
                              void* d_out, int out_size, void* d_ws, size_t ws_size,
                              hipStream_t stream) {
  (void)in_sizes; (void)n_in; (void)out_size; (void)ws_size;
  const float* adj  = (const float*)d_in[0];
  const float* nfda = (const float*)d_in[1];
  const float* qadj = (const float*)d_in[2];
  const float* nfq  = (const float*)d_in[3];
  const int*   cand = (const int*)d_in[4];
  // d_in[5] candidate_adj: unused by the reference
  const float* thr  = (const float*)d_in[6];
  const float* W1da = (const float*)d_in[7];
  const float* W1q  = (const float*)d_in[8];
  const float* W2da = (const float*)d_in[9];
  const float* W2q  = (const float*)d_in[10];

  float* out        = (float*)d_out;
  float* out_end    = out;                                     // [10000]
  float* out_attda2 = out + NDA;                               // [10000*256]
  float* out_attq2  = out + NDA + (size_t)NDA * DD;            // [64*256]
  float* out_scal   = out + NDA + (size_t)NDA * DD + NQ * DD;  // [3]

  float* ws    = (float*)d_ws;                 // needs ~18 MB
  float* buf1  = ws;                           // [NDA*DD] fp32
  __bf16* bufb = (__bf16*)(buf1 + (size_t)NDA * DD);  // [NDA*DD] bf16
  float* p     = (float*)(bufb + (size_t)NDA * DD);
  float* dan   = p; p += (size_t)NC * DD;
  float* danT  = p; p += (size_t)NC * DD;
  float* preda = p; p += (size_t)NC * DD;
  float* qn    = p; p += NQ * DD;
  float* cbuf  = p; p += NQ * NC;
  float* hbuf  = p; p += NQ * DD;
  float* qbuf  = p; p += NQ * DD;
  float* attq1 = p; p += NQ * DD;
  float* tq    = p; p += NQ * DD;
  float* S2    = p; p += NDA;
  float* SAv   = p; p += NDA;
  float* DVv   = p; p += NDA;
  float* msk   = p; p += NDA;
  float* emb   = p; p += (DD + 1);

  dim3 B(256);
  const int n4Da = (NDA * DD) / 4;
  const int nT   = NC * DD;
  const int aggBlocks = (NDA + 31) / 32;       // 313
  auto gemm = [&](const float* A, const float* Bp, void* Cc, int M, int N,
                  int K, int act, int obf) {
    int ntiles = (M / 16) * (N / 16);
    gemm_wmma_kernel<<<(ntiles + 7) / 8, B, 0, stream>>>(A, Bp, Cc, M, N, K, act, obf);
  };

  // ---- layer 1 ----
  gemm(nfda, W1da, bufb, NDA, DD, DIN, 0, 1);                          // T1 (bf16)
  gcn_agg_kernel<<<aggBlocks, B, 0, stream>>>(adj, bufb, buf1, NDA, NDA, 1); // da1
  gather_norm_kernel<<<NC, B, 0, stream>>>(buf1, cand, preda, dan);    // pre_da1, dan
  transpose_kernel<<<(nT + 255) / 256, B, 0, stream>>>(dan, danT, nT);
  gemm(nfq, W1q, tq, NQ, DD, DIN, 0, 0);
  gemm(qadj, tq, qbuf, NQ, DD, NQ, 1, 0);                              // q1
  gather_norm_kernel<<<NQ, B, 0, stream>>>(qbuf, nullptr, nullptr, qn);
  gemm(qn, danT, cbuf, NQ, NC, DD, 0, 0);                              // c1 = qn @ dan^T
  gemm(cbuf, preda, hbuf, NQ, DD, NC, 0, 0);                           // h1_q
  col_norm_add_lrelu_kernel<<<1, B, 0, stream>>>(qbuf, hbuf, attq1);   // att_q1
  lrelu4_kernel<<<(n4Da + 255) / 256, B, 0, stream>>>((const f4*)buf1,
                                                      (f4*)buf1, n4Da); // att_da1

  // ---- layer 2 ----
  gemm(buf1, W2da, bufb, NDA, DD, DD, 0, 1);                           // T2 (bf16)
  gcn_agg_kernel<<<aggBlocks, B, 0, stream>>>(adj, bufb, buf1, NDA, NDA, 1); // da2
  gather_norm_kernel<<<NC, B, 0, stream>>>(buf1, cand, preda, dan);
  transpose_kernel<<<(nT + 255) / 256, B, 0, stream>>>(dan, danT, nT);
  gemm(attq1, W2q, tq, NQ, DD, DD, 0, 0);
  gemm(qadj, tq, qbuf, NQ, DD, NQ, 1, 0);                              // q2
  gather_norm_kernel<<<NQ, B, 0, stream>>>(qbuf, nullptr, nullptr, qn);
  gemm(qn, danT, cbuf, NQ, NC, DD, 0, 0);                              // c2
  gemm(cbuf, preda, hbuf, NQ, DD, NC, 0, 0);                           // h2_q
  col_norm_add_lrelu_kernel<<<1, B, 0, stream>>>(qbuf, hbuf, out_attq2);     // att_q2
  lrelu4_kernel<<<(n4Da + 255) / 256, B, 0, stream>>>((const f4*)buf1,
                                                      (f4*)out_attda2, n4Da);

  // ---- matching score ----
  colmean_kernel<<<1, B, 0, stream>>>(out_attq2, emb);
  end_mask_kernel<<<NDA, B, 0, stream>>>(out_attda2, emb, thr, out_end, msk);

  // ---- graph statistics (re_adj = F F^T, never materialized) ----
  mask_rows_bf_kernel<<<(n4Da + 255) / 256, B, 0, stream>>>((const f4*)out_attda2,
                                                            msk, bufb, n4Da); // F_ bf16
  hipMemsetAsync(S2, 0, NDA * sizeof(float), stream);
  re_rownorm_kernel<<<NDA / 16, B, 0, stream>>>(bufb, S2, NDA);        // row ||.||^2
  gcn_agg_kernel<<<aggBlocks, B, 0, stream>>>(adj, bufb, buf1, NDA, NDA, 0); // G=adj@F_
  row_dot_kernel<<<NDA, B, 0, stream>>>(bufb, buf1, adj, SAv, DVv);
  finalize_kernel<<<1, B, 0, stream>>>(S2, SAv, DVv, msk, out_scal);
}
